// MultiHeadedAttention_77610059039256
// MI455X (gfx1250) — matrix operand order repaired and checked by reading, not compiled, not run on hardware
//
#include <hip/hip_runtime.h>
#include <hip/hip_bf16.h>
#include <stdint.h>

// ---------------------------------------------------------------------------
// LoRA multi-head attention forward for MI455X (gfx1250), wave32 + WMMA + TDM.
//   B=4, T=1024, E=1024, H=16, R=16, DH=64
// Pipeline:
//   1) fuse_w:   W_eff = W + B@A  -> f16           (3x, rank-16 update)
//   2) cvt_x:    X fp32 -> f16
//   3) proj_qkv: Y = Xh @ W_eff.T + bias, WMMA f16, 16x64 tile/wave,
//                double-buffered k-loop -> Qh (scaled 1/32), Kh, Vt
//   4) col_softmax_sum: colsum_j = sum_{i>=j} exp(S[i,j]); store 1/colsum
//                (K tile staged via tensor_load_to_lds; Q loop double-buffered)
//   5) attn_out: P = exp(S)*inv*mask (causal), out = P @ V
//                (Q tile staged via TDM; P transposed C->A layout via LDS)
// ---------------------------------------------------------------------------

typedef _Float16 v8h  __attribute__((ext_vector_type(8)));
typedef _Float16 v16h __attribute__((ext_vector_type(16)));
typedef float    v8f  __attribute__((ext_vector_type(8)));
typedef uint32_t v4u  __attribute__((ext_vector_type(4)));
typedef uint32_t v8u  __attribute__((ext_vector_type(8)));

#define B_  4
#define T_  1024
#define E_  1024
#define H_  16
#define R_  16
#define DH_ 64

static __device__ __forceinline__ v16h cat8(v8h lo, v8h hi) {
  return __builtin_shufflevector(lo, hi, 0,1,2,3,4,5,6,7,8,9,10,11,12,13,14,15);
}

static __device__ __forceinline__ v8f wmma16(v16h a, v16h b, v8f c) {
  // D = A(16x32 f16) x B(32x16 f16) + C(16x16 f32)
  return __builtin_amdgcn_wmma_f32_16x16x32_f16(false, a, false, b,
                                                (short)0, c, false, false);
}

// ---------------------------------------------------------------------------
// TDM: 2D tile load Global -> LDS (f16 elements).
// D# per CDNA5 ISA.
// ---------------------------------------------------------------------------
static __device__ __forceinline__ void tdm_load_2d_f16(
    uint32_t lds_addr, const _Float16* gsrc, uint32_t rows, uint32_t cols,
    uint32_t row_stride) {
  const uint64_t ga = (uint64_t)(uintptr_t)gsrc;
  v4u g0;
  g0[0] = 1u;                                            // count=1
  g0[1] = lds_addr;                                      // LDS byte address
  g0[2] = (uint32_t)ga;                                  // global addr lo
  g0[3] = (uint32_t)((ga >> 32) & 0x1FFFFFFu) | (2u << 30);  // hi | type=2
  v8u g1;
  g1[0] = 0x00010000u;                                   // data_size=1 (2B)
  g1[1] = (cols & 0xFFFFu) << 16;                        // tensor_dim0 lo
  g1[2] = (cols >> 16) | ((rows & 0xFFFFu) << 16);       // dim0 hi | dim1 lo
  g1[3] = (rows >> 16) | ((cols & 0xFFFFu) << 16);       // dim1 hi | tile_dim0
  g1[4] = rows & 0xFFFFu;                                // tile_dim1 (dim2=0)
  g1[5] = row_stride;                                    // dim0_stride lo32
  g1[6] = 0u;
  g1[7] = 0u;
  asm volatile("tensor_load_to_lds %0, %1" :: "s"(g0), "s"(g1) : "memory");
}

static __device__ __forceinline__ uint32_t lds_addr_of(const void* p) {
  // LDS aperture: low 32 bits of a generic pointer are the LDS byte offset.
  return (uint32_t)(uintptr_t)p;
}

// ---------------------------------------------------------------------------
// Kernel 1: W_eff[j,e] = W[j,e] + sum_r Bl[j,r] * A[r,e]   (f16 out)
// ---------------------------------------------------------------------------
__global__ void fuse_w(const float* __restrict__ W, const float* __restrict__ A,
                       const float* __restrict__ Bl, _Float16* __restrict__ out) {
  const int idx = blockIdx.x * blockDim.x + threadIdx.x;  // [0, E*E)
  const int j = idx >> 10;
  const int e = idx & (E_ - 1);
  float acc = W[idx];
#pragma unroll
  for (int r = 0; r < R_; ++r)
    acc += Bl[j * R_ + r] * A[r * E_ + e];
  out[idx] = (_Float16)acc;
}

// ---------------------------------------------------------------------------
// Kernel 2: X fp32 -> f16
// ---------------------------------------------------------------------------
__global__ void cvt_x(const float* __restrict__ X, _Float16* __restrict__ Xh,
                      int n) {
  const int i = blockIdx.x * blockDim.x + threadIdx.x;
  if (i < n) Xh[i] = (_Float16)X[i];
}

// ---------------------------------------------------------------------------
// Kernel 3: QKV projection GEMM. One wave per 16x64 output tile (4 N-tiles,
// one head's columns). Explicit 2-deep software pipeline (ping-pong register
// sets) so next k-step's loads are in flight during current WMMAs.
// ---------------------------------------------------------------------------
static __device__ __forceinline__ void load_ab(const _Float16* arow,
                                               const _Float16* brow, int e0,
                                               v16h& a, v16h b[4]) {
  a = cat8(*(const v8h*)(arow + e0), *(const v8h*)(arow + e0 + 16));
#pragma unroll
  for (int c = 0; c < 4; ++c) {
    const _Float16* bc = brow + (size_t)(16 * c) * E_ + e0;
    b[c] = cat8(*(const v8h*)(bc), *(const v8h*)(bc + 16));
  }
}

__global__ __launch_bounds__(256) void proj_qkv(
    const _Float16* __restrict__ Xh,
    const _Float16* __restrict__ WeQ, const _Float16* __restrict__ WeK,
    const _Float16* __restrict__ WeV,
    const float* __restrict__ bq, const float* __restrict__ bk,
    const float* __restrict__ bv,
    _Float16* __restrict__ Qh, _Float16* __restrict__ Kh,
    _Float16* __restrict__ Vt) {
  const int lane = threadIdx.x & 31;
  const int w    = blockIdx.x * 8 + (threadIdx.x >> 5);  // 12288 waves
  const int proj = w / (256 * 16);
  const int rem  = w % (256 * 16);
  const int m0   = (rem / 16) * 16;  // row tile base in [0, B*T)
  const int j0   = (rem % 16) * 64;  // col tile base (64-aligned -> one head)
  const int g    = lane >> 4;
  const int ln   = lane & 15;

  const _Float16* Wp = (proj == 0) ? WeQ : (proj == 1) ? WeK : WeV;
  const float*    bp = (proj == 0) ? bq  : (proj == 1) ? bk  : bv;

  const _Float16* arow = Xh + (size_t)(m0 + ln) * E_ + 8 * g;
  const _Float16* brow = Wp + (size_t)(j0 + ln) * E_ + 8 * g;

  v8f zero = {};
  v8f acc[4];
#pragma unroll
  for (int c = 0; c < 4; ++c) acc[c] = zero;

  v16h aP, bP[4], aQ2, bQ2[4];
  load_ab(arow, brow, 0, aP, bP);
#pragma unroll 1
  for (int e0 = 0; e0 < E_; e0 += 64) {
    load_ab(arow, brow, e0 + 32, aQ2, bQ2);  // in flight during WMMAs below
#pragma unroll
    for (int c = 0; c < 4; ++c) acc[c] = wmma16(aP, bP[c], acc[c]);
    if (e0 + 64 < E_) load_ab(arow, brow, e0 + 64, aP, bP);
#pragma unroll
    for (int c = 0; c < 4; ++c) acc[c] = wmma16(aQ2, bQ2[c], acc[c]);
  }

  const int h = j0 >> 6;  // wave-uniform head
#pragma unroll
  for (int c = 0; c < 4; ++c) {
    const int dh = 16 * c + ln;       // column within head
    const float bb = bp[j0 + 16 * c + ln];
#pragma unroll
    for (int r = 0; r < 8; ++r) {
      const int ig = m0 + r + 8 * g;  // global row in [0, B*T)
      const int b  = ig >> 10;
      const int t  = ig & 1023;
      const float v = acc[c][r] + bb;
      if (proj == 0) {
        Qh[((size_t)(b * H_ + h) * T_ + t) * DH_ + dh] = (_Float16)(v * 0.03125f);
      } else if (proj == 1) {
        Kh[((size_t)(b * H_ + h) * T_ + t) * DH_ + dh] = (_Float16)v;
      } else {
        Vt[((size_t)(b * H_ + h) * DH_ + dh) * T_ + t] = (_Float16)v;
      }
    }
  }
}

// ---------------------------------------------------------------------------
// Kernel 4: column softmax denominators (softmax over the QUERY axis).
// One wave per (bh, 16-column block); K tile staged to LDS via TDM; Q tile
// loads double-buffered against the WMMA+exp work.
// inv[bh*T + j] = 1 / sum_{i>=j} exp(S[i,j])
// ---------------------------------------------------------------------------
__global__ __launch_bounds__(256) void col_softmax_sum(
    const _Float16* __restrict__ Qh, const _Float16* __restrict__ Kh,
    float* __restrict__ inv) {
  __shared__ _Float16 kst[8][16 * 64];  // 16 KB: one K tile per wave
  const int lane = threadIdx.x & 31;
  const int wv   = threadIdx.x >> 5;
  const int w    = blockIdx.x * 8 + wv;  // 4096 waves
  const int bh   = w >> 6;
  const int j0   = (w & 63) << 4;
  const int g    = lane >> 4;
  const int ln   = lane & 15;

  const _Float16* Q = Qh + (size_t)bh * T_ * DH_;
  const _Float16* K = Kh + (size_t)bh * T_ * DH_;

  // TDM: stage K rows [j0, j0+16) x DH into this wave's LDS region.
  tdm_load_2d_f16(lds_addr_of(&kst[wv][0]), K + (size_t)j0 * DH_, 16, DH_, DH_);
  __builtin_amdgcn_s_wait_tensorcnt(0);

  // B = K^T tiles (d in [0,64)) read from LDS in 16-bit B layout.
  const _Float16* krow = &kst[wv][ln * DH_ + 8 * g];
  const v16h b0 = cat8(*(const v8h*)(krow +  0), *(const v8h*)(krow + 16));
  const v16h b1 = cat8(*(const v8h*)(krow + 32), *(const v8h*)(krow + 48));

  auto loadq = [&](int i0, v16h& a0, v16h& a1) {
    const _Float16* qrow = Q + (size_t)(i0 + ln) * DH_ + 8 * g;
    a0 = cat8(*(const v8h*)(qrow +  0), *(const v8h*)(qrow + 16));
    a1 = cat8(*(const v8h*)(qrow + 32), *(const v8h*)(qrow + 48));
  };

  float csum = 0.f;
  const int jg = j0 + ln;
  v16h c0, c1, n0, n1;
  loadq(j0, c0, c1);
#pragma unroll 1
  for (int i0 = j0; i0 < T_; i0 += 16) {
    if (i0 + 16 < T_) loadq(i0 + 16, n0, n1);  // in flight during WMMA+exp
    v8f s = {};
    s = wmma16(c0, b0, s);
    s = wmma16(c1, b1, s);
#pragma unroll
    for (int r = 0; r < 8; ++r) {
      const int ig = i0 + r + 8 * g;
      csum += (ig >= jg) ? __expf(s[r]) : 0.f;  // causal: j>i masked
    }
    c0 = n0;
    c1 = n1;
  }
  csum += __shfl_xor(csum, 16, 32);  // combine M halves for same column
  if (lane < 16) inv[(size_t)bh * T_ + j0 + lane] = 1.0f / csum;
}

// ---------------------------------------------------------------------------
// Kernel 5: output. One wave per (bh, 16-row block).
//   P[i,j] = (i>=j) ? exp(S[i,j]) * inv[j] * mask[b,j] : 0
//   out[i,d] = sum_j P[i,j] * V[j,d]
// Q tile staged via TDM; S recomputed with WMMA; C->A transpose via LDS.
// ---------------------------------------------------------------------------
__global__ __launch_bounds__(256) void attn_out(
    const _Float16* __restrict__ Qh, const _Float16* __restrict__ Kh,
    const _Float16* __restrict__ Vt, const float* __restrict__ inv,
    const float* __restrict__ mask, float* __restrict__ out) {
  __shared__ _Float16 smem[8][512 + 1024];  // per wave: P(16x32) + Qs(16x64)
  const int lane = threadIdx.x & 31;
  const int wv   = threadIdx.x >> 5;
  const int w    = blockIdx.x * 8 + wv;  // 4096 waves
  const int bh   = w >> 6;
  const int i0   = (w & 63) << 4;
  const int g    = lane >> 4;
  const int ln   = lane & 15;
  const int b    = bh >> 4;
  const int h    = bh & 15;

  const _Float16* Q    = Qh + (size_t)bh * T_ * DH_;
  const _Float16* K    = Kh + (size_t)bh * T_ * DH_;
  const _Float16* V    = Vt + (size_t)bh * DH_ * T_;
  const float*    invc = inv + (size_t)bh * T_;
  const float*    mrow = mask + (size_t)b * T_;
  _Float16* P  = &smem[wv][0];
  _Float16* Qs = &smem[wv][512];

  // TDM: stage Q rows [i0, i0+16) x DH into LDS, then read A layout.
  tdm_load_2d_f16(lds_addr_of(Qs), Q + (size_t)i0 * DH_, 16, DH_, DH_);
  __builtin_amdgcn_s_wait_tensorcnt(0);
  const _Float16* qrow = Qs + ln * DH_ + 8 * g;
  const v16h aq0 = cat8(*(const v8h*)(qrow +  0), *(const v8h*)(qrow + 16));
  const v16h aq1 = cat8(*(const v8h*)(qrow + 32), *(const v8h*)(qrow + 48));

  v8f zero = {};
  v8f acc[4];
#pragma unroll
  for (int c = 0; c < 4; ++c) acc[c] = zero;

#pragma unroll 1
  for (int j0p = 0; j0p <= i0; j0p += 32) {
    if (j0p + 32 <= i0)  // prefetch next j-block's K rows (global_prefetch_b8)
      __builtin_prefetch(K + (size_t)(j0p + 32 + lane) * DH_, 0, 0);
#pragma unroll
    for (int c2 = 0; c2 < 2; ++c2) {
      const int j0 = j0p + 16 * c2;
      if (j0 <= i0) {  // wave-uniform branch: EXEC stays all-ones
        const _Float16* krow = K + (size_t)(j0 + ln) * DH_ + 8 * g;
        const v16h bk0 = cat8(*(const v8h*)(krow +  0), *(const v8h*)(krow + 16));
        const v16h bk1 = cat8(*(const v8h*)(krow + 32), *(const v8h*)(krow + 48));
        v8f s = {};
        s = wmma16(aq0, bk0, s);
        s = wmma16(aq1, bk1, s);
        const int jg = j0 + ln;
        const float scl = invc[jg] * mrow[jg];
#pragma unroll
        for (int r = 0; r < 8; ++r) {
          const int ig = i0 + r + 8 * g;
          const float pe = (ig >= jg) ? __expf(s[r]) * scl : 0.f;
          P[(r + 8 * g) * 32 + 16 * c2 + ln] = (_Float16)pe;  // ds_store_b16
        }
      } else {  // fully above the diagonal: zero tile
#pragma unroll
        for (int r = 0; r < 8; ++r)
          P[(r + 8 * g) * 32 + 16 * c2 + ln] = (_Float16)0.f;
      }
    }
    // LDS in-order within a wave; explicit fence for safety.
    asm volatile("s_wait_dscnt 0" ::: "memory");

    // Reload P in 16-bit A layout: lane(m=ln,g) chunks at j=8g and 16+8g.
    const _Float16* prow = P + ln * 32 + 8 * g;
    const v16h ap = cat8(*(const v8h*)(prow + 0), *(const v8h*)(prow + 16));

    // out += P(16x32) @ V(32x64): 4 N-tiles against transposed V.
#pragma unroll
    for (int c4 = 0; c4 < 4; ++c4) {
      const _Float16* vrow = V + (size_t)(16 * c4 + ln) * T_ + j0p + 8 * g;
      const v16h bvv = cat8(*(const v8h*)(vrow + 0), *(const v8h*)(vrow + 16));
      acc[c4] = wmma16(ap, bvv, acc[c4]);
    }
  }

  // Store out [B,T,E] fp32: row = b*T + t, col = h*64 + d.
#pragma unroll
  for (int c4 = 0; c4 < 4; ++c4) {
#pragma unroll
    for (int r = 0; r < 8; ++r) {
      const int t = i0 + r + 8 * g;
      const int d = 16 * c4 + ln;
      out[(size_t)(b * T_ + t) * E_ + h * DH_ + d] = acc[c4][r];
    }
  }
}

// ---------------------------------------------------------------------------
extern "C" void kernel_launch(void* const* d_in, const int* in_sizes, int n_in,
                              void* d_out, int out_size, void* d_ws,
                              size_t ws_size, hipStream_t stream) {
  const float* X    = (const float*)d_in[0];
  const float* mask = (const float*)d_in[1];
  const float* Wq   = (const float*)d_in[2];
  const float* bq   = (const float*)d_in[3];
  const float* Wk   = (const float*)d_in[4];
  const float* bk   = (const float*)d_in[5];
  const float* Wv   = (const float*)d_in[6];
  const float* bv   = (const float*)d_in[7];
  const float* Aq   = (const float*)d_in[8];
  const float* Bq   = (const float*)d_in[9];
  const float* Ak   = (const float*)d_in[10];
  const float* Bk   = (const float*)d_in[11];
  const float* Av   = (const float*)d_in[12];
  const float* Bv   = (const float*)d_in[13];
  float* out = (float*)d_out;

  char* ws = (char*)d_ws;
  const size_t MB = 1024 * 1024;
  _Float16* WeQ = (_Float16*)(ws);                 // 2 MB
  _Float16* WeK = (_Float16*)(ws + 2 * MB);        // 2 MB
  _Float16* WeV = (_Float16*)(ws + 4 * MB);        // 2 MB
  _Float16* Xh  = (_Float16*)(ws + 6 * MB);        // 8 MB
  _Float16* Qh  = (_Float16*)(ws + 14 * MB);       // 8 MB
  _Float16* Kh  = (_Float16*)(ws + 22 * MB);       // 8 MB
  _Float16* Vt  = (_Float16*)(ws + 30 * MB);       // 8 MB
  float*    inv = (float*)(ws + 38 * MB);          // 256 KB

  fuse_w<<<(E_ * E_) / 256, 256, 0, stream>>>(Wq, Aq, Bq, WeQ);
  fuse_w<<<(E_ * E_) / 256, 256, 0, stream>>>(Wk, Ak, Bk, WeK);
  fuse_w<<<(E_ * E_) / 256, 256, 0, stream>>>(Wv, Av, Bv, WeV);
  cvt_x<<<(B_ * T_ * E_) / 256, 256, 0, stream>>>(X, Xh, B_ * T_ * E_);
  // 3 proj * 256 M-tiles * 16 N64-tiles = 12288 waves / 8 per block
  proj_qkv<<<1536, 256, 0, stream>>>(Xh, WeQ, WeK, WeV, bq, bk, bv, Qh, Kh, Vt);
  // 64 (b,h) * 64 column blocks = 4096 waves
  col_softmax_sum<<<512, 256, 0, stream>>>(Qh, Kh, inv);
  // 64 (b,h) * 64 row blocks = 4096 waves
  attn_out<<<512, 256, 0, stream>>>(Qh, Kh, Vt, inv, mask, out);
}
